// MultiHeadSelfAttention_10995116277808
// MI455X (gfx1250) — matrix operand order repaired and checked by reading, not compiled, not run on hardware
//
#include <hip/hip_runtime.h>
#include <hip/hip_bf16.h>
#include <stdint.h>

typedef __attribute__((ext_vector_type(16))) _Float16 v16h;
typedef __attribute__((ext_vector_type(8)))  _Float16 h8;
typedef __attribute__((ext_vector_type(4)))  _Float16 h4;
typedef __attribute__((ext_vector_type(8)))  float    v8f;
typedef __attribute__((ext_vector_type(4)))  float    f4;
typedef __attribute__((ext_vector_type(4)))  unsigned int u32x4;
typedef __attribute__((ext_vector_type(8)))  int      i32x8;
typedef __attribute__((ext_vector_type(4)))  int      i32x4;

union F16x16 { v16h v; h8 h[2]; };

#define BATCH  4
#define SEQ    2048
#define DMODEL 1024
#define NHEADS 16
#define HDIM   64

static __device__ __forceinline__ v8f wmma_f16(v16h a, v16h b, v8f c) {
  return __builtin_amdgcn_wmma_f32_16x16x32_f16(false, a, false, b, (short)0, c,
                                                false, false);
}

// ---------------------------------------------------------------------------
// 16-lane row reductions. DPP16 butterfly: after xor1/xor2 quad-perms the
// value is quad-uniform, so half-mirror / mirror complete the 16-lane tree
// without touching the LDS unit (vs __shfl_xor -> ds_bpermute_b32).
// ---------------------------------------------------------------------------
#if __has_builtin(__builtin_amdgcn_mov_dpp)
template <int C>
static __device__ __forceinline__ float dppf(float x) {
  return __builtin_bit_cast(
      float, __builtin_amdgcn_mov_dpp(__builtin_bit_cast(int, x), C, 0xf, 0xf,
                                      true));
}
static __device__ __forceinline__ float red16_max(float x) {
  x = fmaxf(x, dppf<0xB1>(x));   // quad_perm [1,0,3,2]  (xor 1)
  x = fmaxf(x, dppf<0x4E>(x));   // quad_perm [2,3,0,1]  (xor 2)
  x = fmaxf(x, dppf<0x141>(x));  // row_half_mirror
  x = fmaxf(x, dppf<0x140>(x));  // row_mirror
  return x;
}
static __device__ __forceinline__ float red16_sum(float x) {
  x += dppf<0xB1>(x);
  x += dppf<0x4E>(x);
  x += dppf<0x141>(x);
  x += dppf<0x140>(x);
  return x;
}
#else
static __device__ __forceinline__ float red16_max(float x) {
  x = fmaxf(x, __shfl_xor(x, 1, 32));
  x = fmaxf(x, __shfl_xor(x, 2, 32));
  x = fmaxf(x, __shfl_xor(x, 4, 32));
  x = fmaxf(x, __shfl_xor(x, 8, 32));
  return x;
}
static __device__ __forceinline__ float red16_sum(float x) {
  x += __shfl_xor(x, 1, 32);
  x += __shfl_xor(x, 2, 32);
  x += __shfl_xor(x, 4, 32);
  x += __shfl_xor(x, 8, 32);
  return x;
}
#endif

// ---------------------------------------------------------------------------
// Tensor Data Mover: 2-D f16 tile (tile_w x tile_h, row stride stride_elems)
// from global to LDS, with LDS row padding via the D# pad fields.
// D# layout per CDNA5 ISA. Issued by one full wave.
// This toolchain exposes the 6-arg builtin (extra i32x8 group, zero-filled;
// our tensors are 2-D with tile_dim2=0 so the extra groups are unused).
// ---------------------------------------------------------------------------
#define TDM_OK __has_builtin(__builtin_amdgcn_tensor_load_to_lds)
#if TDM_OK
static __device__ __forceinline__ void tdm_load_2d_f16(
    unsigned lds_off, const _Float16* gaddr, unsigned tile_w, unsigned tile_h,
    unsigned stride_elems, unsigned pad_int_code, unsigned pad_amt_code) {
  unsigned long long ga = (unsigned long long)(size_t)gaddr;
  u32x4 g0;
  g0[0] = 1u;                                   // count=1 (valid), user mode
  g0[1] = lds_off;                              // lds_addr
  g0[2] = (unsigned)ga;                         // global_addr[31:0]
  g0[3] = (unsigned)((ga >> 32) & 0x01ffffffull) | (2u << 30);  // addr hi,type=2
  i32x8 g1;
  g1[0] = (int)((1u << 16)                      // data_size = 2 bytes
                | (1u << 20)                    // pad_enable
                | (pad_int_code << 22)          // pad interval (DWORDs, log2-1)
                | (pad_amt_code << 25));        // pad amount   (DWORDs-1)
  g1[1] = (int)(tile_w << 16);                  // tensor_dim0[15:0]
  g1[2] = (int)(tile_h << 16);                  // tensor_dim1[15:0]
  g1[3] = (int)(tile_w << 16);                  // tile_dim0
  g1[4] = (int)tile_h;                          // tile_dim1 (tile_dim2 = 0)
  g1[5] = (int)stride_elems;                    // tensor_dim0_stride[31:0]
  g1[6] = 0;
  g1[7] = 0;
  i32x4 z4 = {0, 0, 0, 0};
  i32x8 z8 = {0, 0, 0, 0, 0, 0, 0, 0};
  __builtin_amdgcn_tensor_load_to_lds(g0, g1, z4, z4, z8, 0);
}
#endif

// ---------------------------------------------------------------------------
// f32 -> f16 conversion, vectorized by 4
// ---------------------------------------------------------------------------
__global__ void cvt_f32_f16(const float* __restrict__ in,
                            _Float16* __restrict__ out, int n4) {
  int i = blockIdx.x * blockDim.x + threadIdx.x;
  if (i >= n4) return;
  f4 v = ((const f4*)in)[i];
  h4 o = { (_Float16)v[0], (_Float16)v[1], (_Float16)v[2], (_Float16)v[3] };
  ((h4*)out)[i] = o;
}

// ---------------------------------------------------------------------------
// C[M,N] = A[M,K] * Bw[N,K]^T (f16 in, f32 acc). Block tile 128x256, K-step
// 32. 8 waves, each 64x64 (16 accum tiles, 16 WMMA / k-step). Tiles staged by
// the Tensor Data Mover into double-buffered LDS; DMA of tile k+1 overlaps
// WMMA on tile k; sync = s_wait_tensorcnt 0 (issuing wave) + barrier.
// ---------------------------------------------------------------------------
template <bool OUT_HALF>
__global__ __launch_bounds__(256) void gemm_wmma(
    const _Float16* __restrict__ A, const _Float16* __restrict__ Bw,
    float* __restrict__ Cf, _Float16* __restrict__ Ch, int M, int N, int K) {
  __shared__ _Float16 As[2][128][40];   // [buf][row][k] pad 8 f16 (80B rows)
  __shared__ _Float16 Bs[2][256][40];   // [buf][n][k]

  const int tid  = threadIdx.x;
  const int lane = tid & 31, wave = tid >> 5;
  const int half = lane >> 4, l16 = lane & 15;
  const int bm = blockIdx.y * 128, bn = blockIdx.x * 256;
  const int wm = (wave >> 2) * 64, wn = (wave & 3) * 64;

  v8f acc[4][4];
#pragma unroll
  for (int i = 0; i < 4; ++i)
#pragma unroll
    for (int j = 0; j < 4; ++j) acc[i][j] = (v8f){};

#if TDM_OK
  // 32 f16 per row = 16 DWORDs -> interval code 3; pad 4 DWORDs -> code 3.
  const unsigned ldsA[2] = {(unsigned)(size_t)&As[0][0][0],
                            (unsigned)(size_t)&As[1][0][0]};
  const unsigned ldsB[2] = {(unsigned)(size_t)&Bs[0][0][0],
                            (unsigned)(size_t)&Bs[1][0][0]};
  auto stage = [&](int b, int k0) {
    if (wave == 0) {  // wave-uniform: exactly one TDM issue per descriptor
      tdm_load_2d_f16(ldsA[b], &A[(size_t)bm * K + k0], 32, 128, K, 3, 3);
      tdm_load_2d_f16(ldsB[b], &Bw[(size_t)bn * K + k0], 32, 256, K, 3, 3);
    }
  };
  int buf = 0;
  stage(0, 0);
  if (wave == 0) __builtin_amdgcn_s_wait_tensorcnt(0);
  __syncthreads();
  for (int k0 = 0;; k0 += 32) {
    const int nb = buf ^ 1;
    if (k0 + 32 < K) stage(nb, k0 + 32);   // async: overlaps compute below
    F16x16 af[4], bf[4];
#pragma unroll
    for (int mt = 0; mt < 4; ++mt) {
      int r = wm + mt * 16 + l16;
      af[mt].h[0] = *(const h8*)&As[buf][r][half * 8];
      af[mt].h[1] = *(const h8*)&As[buf][r][16 + half * 8];
    }
#pragma unroll
    for (int nt = 0; nt < 4; ++nt) {
      int r = wn + nt * 16 + l16;
      bf[nt].h[0] = *(const h8*)&Bs[buf][r][half * 8];
      bf[nt].h[1] = *(const h8*)&Bs[buf][r][16 + half * 8];
    }
#pragma unroll
    for (int mt = 0; mt < 4; ++mt)
#pragma unroll
      for (int nt = 0; nt < 4; ++nt)
        acc[mt][nt] = wmma_f16(af[mt].v, bf[nt].v, acc[mt][nt]);
    if (k0 + 32 >= K) break;
    if (wave == 0) __builtin_amdgcn_s_wait_tensorcnt(0);
    __syncthreads();
    buf = nb;
  }
#else
  for (int k0 = 0; k0 < K; k0 += 32) {
#pragma unroll
    for (int i = 0; i < 2; ++i) {          // A: 512 x 16B chunks
      int c = tid + i * 256;
      int row = c >> 2, col = (c & 3) << 3;
      *(h8*)&As[0][row][col] = *(const h8*)&A[(size_t)(bm + row) * K + k0 + col];
    }
#pragma unroll
    for (int i = 0; i < 4; ++i) {          // B: 1024 x 16B chunks
      int c = tid + i * 256;
      int row = c >> 2, col = (c & 3) << 3;
      *(h8*)&Bs[0][row][col] = *(const h8*)&Bw[(size_t)(bn + row) * K + k0 + col];
    }
    __syncthreads();
    F16x16 af[4], bf[4];
#pragma unroll
    for (int mt = 0; mt < 4; ++mt) {
      int r = wm + mt * 16 + l16;
      af[mt].h[0] = *(const h8*)&As[0][r][half * 8];
      af[mt].h[1] = *(const h8*)&As[0][r][16 + half * 8];
    }
#pragma unroll
    for (int nt = 0; nt < 4; ++nt) {
      int r = wn + nt * 16 + l16;
      bf[nt].h[0] = *(const h8*)&Bs[0][r][half * 8];
      bf[nt].h[1] = *(const h8*)&Bs[0][r][16 + half * 8];
    }
#pragma unroll
    for (int mt = 0; mt < 4; ++mt)
#pragma unroll
      for (int nt = 0; nt < 4; ++nt)
        acc[mt][nt] = wmma_f16(af[mt].v, bf[nt].v, acc[mt][nt]);
    __syncthreads();
  }
#endif

  // C/D layout: lane holds column n = l16, rows m = r + 8*half
#pragma unroll
  for (int mt = 0; mt < 4; ++mt)
#pragma unroll
    for (int nt = 0; nt < 4; ++nt)
#pragma unroll
      for (int r = 0; r < 8; ++r) {
        int gm = bm + wm + mt * 16 + r + 8 * half;
        int gn = bn + wn + nt * 16 + l16;
        if (OUT_HALF)
          Ch[(size_t)gm * N + gn] = (_Float16)acc[mt][nt][r];
        else
          Cf[(size_t)gm * N + gn] = acc[mt][nt][r];
      }
}

// ---------------------------------------------------------------------------
// Causal flash attention. Grid (S/64, H, B), 128 threads (4 waves).
// ---------------------------------------------------------------------------
__global__ __launch_bounds__(128) void attn_wmma(
    const _Float16* __restrict__ qkv, _Float16* __restrict__ outh) {
  __shared__ _Float16 Kt[32][72];      // [key][hd]
  __shared__ _Float16 Vt[64][40];      // transposed: [hd][key]
  __shared__ _Float16 Pw[4][16][40];   // per-wave P staging (C->A relayout)

  const int tid  = threadIdx.x;
  const int lane = tid & 31, wave = tid >> 5;
  const int half = lane >> 4, l16 = lane & 15;
  const int qt = blockIdx.x, h = blockIdx.y, b = blockIdx.z;

  const int q0 = qt * 64 + wave * 16;
  const size_t rs   = 3 * DMODEL;
  const size_t base = (size_t)b * SEQ * rs;

  F16x16 qa[2];
  {
    const size_t qrow = base + (size_t)(q0 + l16) * rs + (size_t)h * HDIM;
#pragma unroll
    for (int j = 0; j < 2; ++j) {
      qa[j].h[0] = *(const h8*)&qkv[qrow + j * 32 + half * 8];
      qa[j].h[1] = *(const h8*)&qkv[qrow + j * 32 + 16 + half * 8];
    }
  }

  v8f o[4];
#pragma unroll
  for (int i = 0; i < 4; ++i) o[i] = (v8f){};
  float mi[8], li[8];
#pragma unroll
  for (int r = 0; r < 8; ++r) { mi[r] = -1e30f; li[r] = 0.f; }

  const int nkb = qt * 2 + 2;
  for (int kb = 0; kb < nkb; ++kb) {
    const int kbase = kb * 32;
#pragma unroll
    for (int i = 0; i < 2; ++i) {
      int c = tid + i * 128;
      int krow = c >> 3, c8 = (c & 7) * 8;
      size_t g = base + (size_t)(kbase + krow) * rs + DMODEL +
                 (size_t)h * HDIM + c8;
      *(h8*)&Kt[krow][c8] = *(const h8*)&qkv[g];
      h8 vv = *(const h8*)&qkv[g + DMODEL];
#pragma unroll
      for (int e = 0; e < 8; ++e) Vt[c8 + e][krow] = vv[e];
    }
    __syncthreads();

    if (kbase <= q0 + 15) {  // wave-uniform: EXEC all-ones for WMMA
      v8f sc[2];
#pragma unroll
      for (int nt = 0; nt < 2; ++nt) {
        sc[nt] = (v8f){};
#pragma unroll
        for (int j = 0; j < 2; ++j) {
          F16x16 kf;
          kf.h[0] = *(const h8*)&Kt[nt * 16 + l16][j * 32 + half * 8];
          kf.h[1] = *(const h8*)&Kt[nt * 16 + l16][j * 32 + 16 + half * 8];
          sc[nt] = wmma_f16(qa[j].v, kf.v, sc[nt]);
        }
      }
      float alpha[8];
#pragma unroll
      for (int r = 0; r < 8; ++r) {
        int qg = q0 + r + 8 * half;
#pragma unroll
        for (int nt = 0; nt < 2; ++nt) {
          int kg = kbase + nt * 16 + l16;
          float s = sc[nt][r] * 0.125f;        // 1/sqrt(64)
          sc[nt][r] = (kg > qg) ? -1e30f : s;
        }
        float t = red16_max(fmaxf(sc[0][r], sc[1][r]));
        float mnew = fmaxf(mi[r], t);
        alpha[r] = __expf(mi[r] - mnew);
        float p0 = __expf(sc[0][r] - mnew);
        float p1 = __expf(sc[1][r] - mnew);
        sc[0][r] = p0; sc[1][r] = p1;
        li[r] = li[r] * alpha[r] + red16_sum(p0 + p1);
        mi[r] = mnew;
      }
#pragma unroll
      for (int ot = 0; ot < 4; ++ot)
#pragma unroll
        for (int r = 0; r < 8; ++r) o[ot][r] *= alpha[r];

#pragma unroll
      for (int nt = 0; nt < 2; ++nt)
#pragma unroll
        for (int r = 0; r < 8; ++r)
          Pw[wave][r + 8 * half][nt * 16 + l16] = (_Float16)sc[nt][r];
      F16x16 pa;
      pa.h[0] = *(const h8*)&Pw[wave][l16][half * 8];
      pa.h[1] = *(const h8*)&Pw[wave][l16][16 + half * 8];
#pragma unroll
      for (int ot = 0; ot < 4; ++ot) {
        F16x16 vf;
        vf.h[0] = *(const h8*)&Vt[ot * 16 + l16][half * 8];
        vf.h[1] = *(const h8*)&Vt[ot * 16 + l16][16 + half * 8];
        o[ot] = wmma_f16(pa.v, vf.v, o[ot]);
      }
    }
    __syncthreads();
  }

#pragma unroll
  for (int ot = 0; ot < 4; ++ot)
#pragma unroll
    for (int r = 0; r < 8; ++r) {
      int sq = q0 + r + 8 * half;
      float v = o[ot][r] / li[r];
      outh[((size_t)b * SEQ + sq) * DMODEL + (size_t)h * HDIM + ot * 16 + l16] =
          (_Float16)v;
    }
}

// ---------------------------------------------------------------------------
extern "C" void kernel_launch(void* const* d_in, const int* in_sizes, int n_in,
                              void* d_out, int out_size, void* d_ws,
                              size_t ws_size, hipStream_t stream) {
  (void)in_sizes; (void)n_in; (void)out_size; (void)ws_size;
  const float* x    = (const float*)d_in[0];
  const float* wqkv = (const float*)d_in[1];
  const float* wout = (const float*)d_in[2];
  float* out = (float*)d_out;

  const size_t MB = 1024ull * 1024ull;
  char* ws = (char*)d_ws;
  _Float16* xh    = (_Float16*)(ws + 0 * MB);   // 16 MB
  _Float16* wqkvh = (_Float16*)(ws + 16 * MB);  //  6 MB
  _Float16* wouth = (_Float16*)(ws + 22 * MB);  //  2 MB
  _Float16* qkvh  = (_Float16*)(ws + 24 * MB);  // 48 MB
  _Float16* atth  = (_Float16*)(ws + 72 * MB);  // 16 MB

  const int M = BATCH * SEQ;  // 8192

  cvt_f32_f16<<<(M * DMODEL / 4 + 255) / 256, 256, 0, stream>>>(
      x, xh, M * DMODEL / 4);
  cvt_f32_f16<<<(3 * DMODEL * DMODEL / 4 + 255) / 256, 256, 0, stream>>>(
      wqkv, wqkvh, 3 * DMODEL * DMODEL / 4);
  cvt_f32_f16<<<(DMODEL * DMODEL / 4 + 255) / 256, 256, 0, stream>>>(
      wout, wouth, DMODEL * DMODEL / 4);

  gemm_wmma<true><<<dim3(3 * DMODEL / 256, M / 128), 256, 0, stream>>>(
      xh, wqkvh, nullptr, qkvh, M, 3 * DMODEL, DMODEL);

  attn_wmma<<<dim3(SEQ / 64, NHEADS, BATCH), 128, 0, stream>>>(qkvh, atth);

  gemm_wmma<false><<<dim3(DMODEL / 256, M / 128), 256, 0, stream>>>(
      atth, wouth, out, nullptr, M, DMODEL, DMODEL);
}
